// Attention_19404662243470
// MI455X (gfx1250) — matrix operand order repaired and checked by reading, not compiled, not run on hardware
//
#include <hip/hip_runtime.h>
#include <cstdint>
#include <cstddef>

// ---------------------------------------------------------------------------
// Attention block for MI455X (gfx1250, wave32, WMMA bf16 16x16x32).
// B=2, DIM=128, H=W=64 (N=4096), HEADS=4, DH=64, INNER=256, QKV=768.
// ---------------------------------------------------------------------------

typedef unsigned short u16;
typedef __attribute__((ext_vector_type(16))) __bf16 bf16x16;
typedef __attribute__((ext_vector_type(8)))  float  f32x8;

union ABFrag { bf16x16 v; uint4 u[2]; };

#define WMMA_BF16(A, Bm, C) \
  __builtin_amdgcn_wmma_f32_16x16x32_bf16(false, (A), false, (Bm), (short)0, (C), false, false)

#define NB 2        // batch
#define DIMC 128
#define NSP 4096    // H*W
#define NHEADS 4
#define DHD 64
#define INNERC 256
#define QKVC 768

__device__ __forceinline__ u16 f2bf(float f) {
  union { float f; uint32_t u; } c; c.f = f;
  uint32_t u = c.u;
  u += 0x7fffu + ((u >> 16) & 1u);          // round-to-nearest-even
  return (u16)(u >> 16);
}
__device__ __forceinline__ float bf2f(u16 s) {
  union { uint32_t u; float f; } c; c.u = ((uint32_t)s) << 16;
  return c.f;
}

// ---------------------------------------------------------------------------
// K0: convert pointwise weights fp32 -> bf16 (row-major kept).
// ---------------------------------------------------------------------------
__global__ void k_cvt_w(const float* __restrict__ qkv_pw, const float* __restrict__ out_pw,
                        u16* __restrict__ wq, u16* __restrict__ wo) {
  int i = blockIdx.x * blockDim.x + threadIdx.x;
  if (i < QKVC * DIMC)  wq[i] = f2bf(qkv_pw[i]);
  if (i < DIMC * INNERC) wo[i] = f2bf(out_pw[i]);
}

// ---------------------------------------------------------------------------
// K1: x + positional encoding (fp32).
// ---------------------------------------------------------------------------
__global__ void k_posadd(const float* __restrict__ x, const float* __restrict__ pe_w,
                         const float* __restrict__ pe_b, float* __restrict__ xp) {
  int i = blockIdx.x * blockDim.x + threadIdx.x;   // over NB*DIMC*NSP
  int n = i & (NSP - 1);
  int bc = i >> 12;
  int c = bc & (DIMC - 1);
  int hh = n >> 6, wc = n & 63;
  float gx = hh * (1.0f / 63.0f);
  float gy = wc * (1.0f / 63.0f);
  xp[i] = x[i] + pe_w[c * 2 + 0] * gx + pe_w[c * 2 + 1] * gy + pe_b[c];
}

// ---------------------------------------------------------------------------
// K2: depthwise 3x3 SAME conv (cross-correlation), bf16 output [b][c][n].
// ---------------------------------------------------------------------------
__global__ void k_dwconv(const float* __restrict__ x, const float* __restrict__ dw,
                         u16* __restrict__ y, int C) {
  int i = blockIdx.x * blockDim.x + threadIdx.x;   // over NB*C*NSP
  int n = i & (NSP - 1);
  int bc = i >> 12;
  int c = bc % C;
  int hh = n >> 6, wc = n & 63;
  const float* xin = x + ((size_t)bc << 12);
  const float* wgt = dw + c * 9;
  float acc = 0.0f;
#pragma unroll
  for (int ky = 0; ky < 3; ++ky) {
    int yy = hh + ky - 1;
    if (yy < 0 || yy > 63) continue;
#pragma unroll
    for (int kx = 0; kx < 3; ++kx) {
      int xx = wc + kx - 1;
      if (xx < 0 || xx > 63) continue;
      acc += xin[yy * 64 + xx] * wgt[ky * 3 + kx];
    }
  }
  y[i] = f2bf(acc);
}

// ---------------------------------------------------------------------------
// K3/K7: bf16 WMMA GEMM.  Out[b][m][n] = sum_k A[m][k] * Bmat[b][k][n]
//   A row-major MxK bf16, Bmat [b][K][Nc] bf16 (n contiguous).
//   One wave per 16x64 output strip.  K is compile-time: K-loop fully
//   unrolls, all fragment loads of a k-step are issued ahead of the
//   4-WMMA chain so the scheduler can overlap VMEM with the matrix pipe.
// ---------------------------------------------------------------------------
template <int K, int OUTBF>
__global__ __launch_bounds__(32)
void k_gemm_bf16(const u16* __restrict__ A, const u16* __restrict__ Bmat,
                 void* __restrict__ Out, int M, int Nc, int mtiles, int n4tiles) {
  int t = blockIdx.x;
  int n4 = t % n4tiles; t /= n4tiles;
  int mt = t % mtiles;
  int b  = t / mtiles;
  int m0 = mt * 16, n0 = n4 * 64;

  int lane = threadIdx.x & 31;
  int hf = (lane >> 4) & 1;
  int lr = lane & 15;

  const u16* Bp = Bmat + (size_t)b * K * Nc;
  f32x8 acc[4];
#pragma unroll
  for (int j = 0; j < 4; ++j) acc[j] = (f32x8){};

  const u16* ap = A + (size_t)(m0 + lr) * K + hf * 8;
#pragma unroll
  for (int k0 = 0; k0 < K; k0 += 32) {
    // issue all loads for this k-step first
    ABFrag a;
    a.u[0] = *(const uint4*)(ap + k0);
    a.u[1] = *(const uint4*)(ap + k0 + 16);
    const u16* bp = Bp + (size_t)(k0 + hf * 8) * Nc + n0 + lr;
    ABFrag bb[4];
#pragma unroll
    for (int j = 0; j < 4; ++j) {
#pragma unroll
      for (int s = 0; s < 16; ++s)
        ((u16*)&bb[j])[s] = bp[(size_t)((s & 7) + (s >> 3) * 16) * Nc + j * 16];
    }
    // then the 4 independent WMMA chains
#pragma unroll
    for (int j = 0; j < 4; ++j)
      acc[j] = WMMA_BF16(a.v, bb[j].v, acc[j]);
  }

#pragma unroll
  for (int j = 0; j < 4; ++j) {
#pragma unroll
    for (int r = 0; r < 8; ++r) {
      int m = m0 + r + hf * 8;
      size_t idx = ((size_t)(b * M + m)) * Nc + n0 + j * 16 + lr;
      if (OUTBF) ((u16*)Out)[idx] = f2bf(acc[j][r]);
      else       ((float*)Out)[idx] = acc[j][r];
    }
  }
}

// ---------------------------------------------------------------------------
// K4: split qkv -> per-head tensors with LayerNorm on q,k.
//   q  bf16 [b][h][n][dh]  (LN'd, pre-scaled by dh^-0.5)
//   kT bf16 [b][h][dh][n]  (LN'd, transposed for QK^T B-frags)
//   v  bf16 [b][h][n][dh]
// channel(head,d) = d*4 + head inside each 256-channel group.
// ---------------------------------------------------------------------------
__global__ void k_qkvperm(const u16* __restrict__ qkv,
                          const float* __restrict__ nqw, const float* __restrict__ nqb,
                          const float* __restrict__ nkw, const float* __restrict__ nkb,
                          u16* __restrict__ q, u16* __restrict__ kT, u16* __restrict__ v) {
  int i = blockIdx.x * blockDim.x + threadIdx.x;   // over NB*NHEADS*NSP
  if (i >= NB * NHEADS * NSP) return;
  int n = i & (NSP - 1);
  int bh = i >> 12;
  int h = bh & (NHEADS - 1);
  int b = bh >> 2;
  const u16* base = qkv + (size_t)b * QKVC * NSP + n;

  float buf[DHD];

  // ---- q ----
#pragma unroll 8
  for (int d = 0; d < DHD; ++d) buf[d] = bf2f(base[(size_t)(d * 4 + h) * NSP]);
  float s = 0.0f;
#pragma unroll 8
  for (int d = 0; d < DHD; ++d) s += buf[d];
  float mu = s * (1.0f / DHD);
  float vs = 0.0f;
#pragma unroll 8
  for (int d = 0; d < DHD; ++d) { float t = buf[d] - mu; vs += t * t; }
  float rstd = rsqrtf(vs * (1.0f / DHD) + 1e-6f);
#pragma unroll 8
  for (int d = 0; d < DHD; ++d) {
    float yv = (buf[d] - mu) * rstd * nqw[h * DHD + d] + nqb[h * DHD + d];
    q[((size_t)bh * NSP + n) * DHD + d] = f2bf(yv * 0.125f);  // fold dh^-0.5
  }

  // ---- k ----
#pragma unroll 8
  for (int d = 0; d < DHD; ++d) buf[d] = bf2f(base[(size_t)(INNERC + d * 4 + h) * NSP]);
  s = 0.0f;
#pragma unroll 8
  for (int d = 0; d < DHD; ++d) s += buf[d];
  mu = s * (1.0f / DHD);
  vs = 0.0f;
#pragma unroll 8
  for (int d = 0; d < DHD; ++d) { float t = buf[d] - mu; vs += t * t; }
  rstd = rsqrtf(vs * (1.0f / DHD) + 1e-6f);
#pragma unroll 8
  for (int d = 0; d < DHD; ++d) {
    float yv = (buf[d] - mu) * rstd * nkw[h * DHD + d] + nkb[h * DHD + d];
    kT[(size_t)bh * DHD * NSP + (size_t)d * NSP + n] = f2bf(yv);
  }

  // ---- v ----
#pragma unroll 8
  for (int d = 0; d < DHD; ++d)
    v[((size_t)bh * NSP + n) * DHD + d] = base[(size_t)(2 * INNERC + d * 4 + h) * NSP];
}

// ---------------------------------------------------------------------------
// K5: flash attention. grid = NB*NHEADS*64 blocks, 128 threads (4 waves).
// Each wave owns 16 query rows; kv processed in tiles of 32 with online
// softmax.  All K and V fragments of the tile are issued up front: the
// S-WMMAs wait only for the (older) K loads, and the V loads drain behind
// the softmax VALU stretch so the P.V WMMAs start with resident data.
// Epilogue: /l, +v skip, head-LN, scatter to [b][256][n] fp32.
// ---------------------------------------------------------------------------
__global__ __launch_bounds__(128)
void k_attn(const u16* __restrict__ qg, const u16* __restrict__ kTg,
            const u16* __restrict__ vg, const float* __restrict__ now,
            const float* __restrict__ nob, float* __restrict__ outp) {
  int blk = blockIdx.x;
  int qblk = blk & 63;
  int bh = blk >> 6;
  int h = bh & (NHEADS - 1);
  int b = bh >> 2;

  int wave = threadIdx.x >> 5;
  int lane = threadIdx.x & 31;
  int hf = (lane >> 4) & 1;
  int lr = lane & 15;
  int q0 = qblk * 64 + wave * 16;

  const u16* Q  = qg  + (size_t)bh * NSP * DHD;
  const u16* KT = kTg + (size_t)bh * DHD * NSP;
  const u16* V  = vg  + (size_t)bh * NSP * DHD;

  __shared__ __align__(16) u16 pbuf[4][16][40];   // per-wave P staging (padded)
  u16 (*pb)[40] = pbuf[wave];

  // Q A-frags (16 rows x 64 d = two 16x32 frags), resident for whole kernel.
  ABFrag qa[2];
  {
    const u16* qrow = Q + (size_t)(q0 + lr) * DHD + hf * 8;
    qa[0].u[0] = *(const uint4*)(qrow);
    qa[0].u[1] = *(const uint4*)(qrow + 16);
    qa[1].u[0] = *(const uint4*)(qrow + 32);
    qa[1].u[1] = *(const uint4*)(qrow + 48);
  }

  float m_i[8], l_i[8];
  f32x8 oacc[4];
#pragma unroll
  for (int r = 0; r < 8; ++r) { m_i[r] = -3.0e38f; l_i[r] = 0.0f; }
#pragma unroll
  for (int d = 0; d < 4; ++d) oacc[d] = (f32x8){};

  for (int kv0 = 0; kv0 < NSP; kv0 += 32) {
    // prefetch next kv tile's K/V rows into the cache hierarchy
    if (kv0 + 32 < NSP) {
      __builtin_prefetch(KT + (size_t)lane * NSP + kv0 + 32, 0, 1);
      __builtin_prefetch(KT + (size_t)(32 + lane) * NSP + kv0 + 32, 0, 1);
      __builtin_prefetch(V + (size_t)(kv0 + 32 + lane) * DHD, 0, 1);
    }

    // ---- issue ALL fragment loads of this tile first (K then V) ----
    ABFrag kf[4];   // [kb*2+nb]: kb = d 32-block, nb = kv 16-block
#pragma unroll
    for (int kb = 0; kb < 2; ++kb)
#pragma unroll
      for (int nb = 0; nb < 2; ++nb) {
        const u16* kp = KT + (size_t)(kb * 32 + hf * 8) * NSP + kv0 + nb * 16 + lr;
#pragma unroll
        for (int s = 0; s < 16; ++s)
          ((u16*)&kf[kb * 2 + nb])[s] = kp[(size_t)((s & 7) + (s >> 3) * 16) * NSP];
      }
    ABFrag vb[4];   // [db]: d 16-block
#pragma unroll
    for (int db = 0; db < 4; ++db) {
      const u16* vp = V + (size_t)(kv0 + hf * 8) * DHD + db * 16 + lr;
#pragma unroll
      for (int s = 0; s < 16; ++s)
        ((u16*)&vb[db])[s] = vp[((s & 7) + (s >> 3) * 16) * DHD];
    }

    // ---- S = Q * K^T (16 x 32 tile, two 16x16 WMMA accumulators) ----
    f32x8 s[2];
#pragma unroll
    for (int nb = 0; nb < 2; ++nb) {
      f32x8 acc = {};
#pragma unroll
      for (int kb = 0; kb < 2; ++kb)
        acc = WMMA_BF16(qa[kb].v, kf[kb * 2 + nb].v, acc);
      s[nb] = acc;
    }

    // ---- online softmax (row reductions across 16-lane halves) ----
    float scale[8];
#pragma unroll
    for (int r = 0; r < 8; ++r) {
      float mx = fmaxf(s[0][r], s[1][r]);
      mx = fmaxf(mx, __shfl_xor(mx, 1));
      mx = fmaxf(mx, __shfl_xor(mx, 2));
      mx = fmaxf(mx, __shfl_xor(mx, 4));
      mx = fmaxf(mx, __shfl_xor(mx, 8));
      float mn = fmaxf(m_i[r], mx);
      float sc = __expf(m_i[r] - mn);
      float p0 = __expf(s[0][r] - mn);
      float p1 = __expf(s[1][r] - mn);
      s[0][r] = p0; s[1][r] = p1;
      float ps = p0 + p1;
      ps += __shfl_xor(ps, 1);
      ps += __shfl_xor(ps, 2);
      ps += __shfl_xor(ps, 4);
      ps += __shfl_xor(ps, 8);
      l_i[r] = l_i[r] * sc + ps;
      m_i[r] = mn;
      scale[r] = sc;
    }
#pragma unroll
    for (int d = 0; d < 4; ++d)
#pragma unroll
      for (int r = 0; r < 8; ++r) oacc[d][r] *= scale[r];

    // ---- P: D-layout -> bf16 A-layout via wave-private LDS ----
#pragma unroll
    for (int nb = 0; nb < 2; ++nb)
#pragma unroll
      for (int r = 0; r < 8; ++r)
        pb[r + hf * 8][nb * 16 + lr] = f2bf(s[nb][r]);

    ABFrag pa;
    pa.u[0] = *(const uint4*)&pb[lr][hf * 8];
    pa.u[1] = *(const uint4*)&pb[lr][16 + hf * 8];

    // ---- O += P * V (four 16x16 d-blocks) ----
#pragma unroll
    for (int db = 0; db < 4; ++db)
      oacc[db] = WMMA_BF16(pa.v, vb[db].v, oacc[db]);
  }

  // ---- epilogue: /l, +v skip, head-LN, scatter-out ----
  float wv[4], bvv[4];
#pragma unroll
  for (int db = 0; db < 4; ++db) {
    int d = db * 16 + lr;
    wv[db] = now[h * DHD + d];
    bvv[db] = nob[h * DHD + d];
  }

  float o2[4][8];
#pragma unroll
  for (int r = 0; r < 8; ++r) {
    float inv = 1.0f / l_i[r];
    int row = q0 + r + hf * 8;
#pragma unroll
    for (int db = 0; db < 4; ++db) {
      int d = db * 16 + lr;
      o2[db][r] = oacc[db][r] * inv + bf2f(V[(size_t)row * DHD + d]);
    }
  }

#pragma unroll
  for (int r = 0; r < 8; ++r) {
    float s = o2[0][r] + o2[1][r] + o2[2][r] + o2[3][r];
    s += __shfl_xor(s, 1);
    s += __shfl_xor(s, 2);
    s += __shfl_xor(s, 4);
    s += __shfl_xor(s, 8);
    float mu = s * (1.0f / DHD);
    float vs = 0.0f;
#pragma unroll
    for (int db = 0; db < 4; ++db) { float t = o2[db][r] - mu; vs += t * t; }
    vs += __shfl_xor(vs, 1);
    vs += __shfl_xor(vs, 2);
    vs += __shfl_xor(vs, 4);
    vs += __shfl_xor(vs, 8);
    float rstd = rsqrtf(vs * (1.0f / DHD) + 1e-6f);
    int row = q0 + r + hf * 8;
#pragma unroll
    for (int db = 0; db < 4; ++db) {
      int d = db * 16 + lr;
      float yv = (o2[db][r] - mu) * rstd * wv[db] + bvv[db];
      outp[((size_t)(b * INNERC + d * 4 + h)) * NSP + row] = yv;
    }
  }
}

// ---------------------------------------------------------------------------
// K8: LayerNorm2d over channels (128) at each (b, n), fp32 out.
// ---------------------------------------------------------------------------
__global__ void k_ln2d(const float* __restrict__ og, const float* __restrict__ lw,
                       const float* __restrict__ lb, float* __restrict__ out) {
  int i = blockIdx.x * blockDim.x + threadIdx.x;   // over NB*NSP
  if (i >= NB * NSP) return;
  int n = i & (NSP - 1);
  int b = i >> 12;
  const float* p = og + (size_t)b * DIMC * NSP + n;
  float s = 0.0f, s2 = 0.0f;
#pragma unroll 8
  for (int c = 0; c < DIMC; ++c) {
    float v = p[(size_t)c * NSP];
    s += v; s2 += v * v;
  }
  float mu = s * (1.0f / DIMC);
  float var = s2 * (1.0f / DIMC) - mu * mu;
  var = fmaxf(var, 0.0f);
  float rstd = rsqrtf(var + 1e-6f);
  float* o = out + (size_t)b * DIMC * NSP + n;
#pragma unroll 8
  for (int c = 0; c < DIMC; ++c)
    o[(size_t)c * NSP] = (p[(size_t)c * NSP] - mu) * rstd * lw[c] + lb[c];
}

// ---------------------------------------------------------------------------
// launcher
// ---------------------------------------------------------------------------
extern "C" void kernel_launch(void* const* d_in, const int* in_sizes, int n_in,
                              void* d_out, int out_size, void* d_ws, size_t ws_size,
                              hipStream_t stream) {
  (void)in_sizes; (void)n_in; (void)out_size; (void)ws_size;

  const float* x      = (const float*)d_in[0];
  const float* pe_w   = (const float*)d_in[1];
  const float* pe_b   = (const float*)d_in[2];
  const float* qkv_dw = (const float*)d_in[3];
  const float* qkv_pw = (const float*)d_in[4];
  const float* out_dw = (const float*)d_in[5];
  const float* out_pw = (const float*)d_in[6];
  const float* nq_w   = (const float*)d_in[7];
  const float* nq_b   = (const float*)d_in[8];
  const float* nk_w   = (const float*)d_in[9];
  const float* nk_b   = (const float*)d_in[10];
  const float* no_w   = (const float*)d_in[11];
  const float* no_b   = (const float*)d_in[12];
  const float* ln_w   = (const float*)d_in[13];
  const float* ln_b   = (const float*)d_in[14];

  char* ws = (char*)d_ws;
  size_t o_wq  = 0;                                           // 768*128 bf16
  size_t o_wo  = o_wq  + (size_t)QKVC * DIMC * 2;             // 128*256 bf16
  size_t o_xp  = o_wo  + (size_t)DIMC * INNERC * 2;           // x+pos fp32
  size_t o_y   = o_xp  + (size_t)NB * DIMC * NSP * 4;         // dw out bf16
  size_t o_qkv = o_y   + (size_t)NB * DIMC * NSP * 2;         // qkv bf16
  size_t o_q   = o_qkv + (size_t)NB * QKVC * NSP * 2;         // q bf16
  size_t o_kT  = o_q   + (size_t)NB * NHEADS * NSP * DHD * 2; // kT bf16
  size_t o_v   = o_kT  + (size_t)NB * NHEADS * DHD * NSP * 2; // v bf16
  size_t o_ao  = o_v   + (size_t)NB * NHEADS * NSP * DHD * 2; // attn out fp32
  size_t o_y2  = o_ao  + (size_t)NB * INNERC * NSP * 4;       // dw2 out bf16
  size_t o_og  = o_y2  + (size_t)NB * INNERC * NSP * 2;       // gemm2 out fp32

  u16*   wq   = (u16*)(ws + o_wq);
  u16*   wo   = (u16*)(ws + o_wo);
  float* xp   = (float*)(ws + o_xp);
  u16*   ybf  = (u16*)(ws + o_y);
  u16*   qkvb = (u16*)(ws + o_qkv);
  u16*   qb   = (u16*)(ws + o_q);
  u16*   kTb  = (u16*)(ws + o_kT);
  u16*   vb   = (u16*)(ws + o_v);
  float* ao   = (float*)(ws + o_ao);
  u16*   y2   = (u16*)(ws + o_y2);
  float* og   = (float*)(ws + o_og);

  // 0) weight conversion
  k_cvt_w<<<(QKVC * DIMC + 255) / 256, 256, 0, stream>>>(qkv_pw, out_pw, wq, wo);

  // 1) x + pos
  k_posadd<<<(NB * DIMC * NSP) / 256, 256, 0, stream>>>(x, pe_w, pe_b, xp);

  // 2) depthwise 3x3 (qkv path)
  k_dwconv<<<(NB * DIMC * NSP) / 256, 256, 0, stream>>>(xp, qkv_dw, ybf, DIMC);

  // 3) qkv pointwise GEMM: 768x128 @ 128x4096 per batch (WMMA, 16x64/wave)
  k_gemm_bf16<DIMC, 1><<<NB * (QKVC / 16) * (NSP / 64), 32, 0, stream>>>(
      wq, ybf, (void*)qkvb, QKVC, NSP, QKVC / 16, NSP / 64);

  // 4) split heads + LN(q,k) + transpose k
  k_qkvperm<<<(NB * NHEADS * NSP) / 256, 256, 0, stream>>>(
      qkvb, nq_w, nq_b, nk_w, nk_b, qb, kTb, vb);

  // 5) flash attention + skip + head LN (WMMA)
  k_attn<<<NB * NHEADS * 64, 128, 0, stream>>>(qb, kTb, vb, no_w, no_b, ao);

  // 6) depthwise 3x3 (out path)
  k_dwconv<<<(NB * INNERC * NSP) / 256, 256, 0, stream>>>(ao, out_dw, y2, INNERC);

  // 7) out pointwise GEMM: 128x256 @ 256x4096 per batch (WMMA, 16x64/wave)
  k_gemm_bf16<INNERC, 0><<<NB * (DIMC / 16) * (NSP / 64), 32, 0, stream>>>(
      wo, y2, (void*)og, DIMC, NSP, DIMC / 16, NSP / 64);

  // 8) channel LayerNorm -> d_out
  k_ln2d<<<(NB * NSP) / 256, 256, 0, stream>>>(og, ln_w, ln_b, (float*)d_out);
}
